// OptimizedAttention_57655640982122
// MI455X (gfx1250) — matrix operand order repaired and checked by reading, not compiled, not run on hardware
//
#include <hip/hip_runtime.h>
#include <hip/hip_bf16.h>

typedef __bf16 bf16_t;
typedef __attribute__((ext_vector_type(16))) __bf16 v16bf;
typedef __attribute__((ext_vector_type(8)))  float  v8f;
typedef __attribute__((ext_vector_type(4)))  unsigned int u32x4;
typedef __attribute__((ext_vector_type(8)))  int  i32x8;
typedef __attribute__((ext_vector_type(4)))  int  i32x4;

#define NHEADS 16
#define SEQ    2048
#define DMODEL 2048
#define HDIM   128
#define QKVDIM (3 * DMODEL)
#define BATCH  2
#define MROWS  (BATCH * SEQ)   // 4096

#if defined(__gfx1250__) && __has_builtin(__builtin_amdgcn_tensor_load_to_lds)
#define HAVE_TDM 1
#else
#define HAVE_TDM 0
#endif

__device__ __forceinline__ v8f wmma_bf16(v16bf a, v16bf b, v8f c) {
  return __builtin_amdgcn_wmma_f32_16x16x32_bf16(
      /*neg_a=*/false, a, /*neg_b=*/false, b,
      /*c_mod=*/(short)0, c, /*reuse_a=*/false, /*reuse_b=*/false);
}

// A-matrix fragment (16x32 bf16) from row-major LDS tile [rows][ld].
// lane l: M = l&15 ; K = k0 + (l&16 ? 16 : 0) + i.
__device__ __forceinline__ v16bf load_a_frag(const bf16_t* s, int ld,
                                             int row0, int k0, int lane) {
  int m  = lane & 15;
  int kb = k0 + ((lane & 16) ? 16 : 0);
  const bf16_t* p = s + (row0 + m) * ld + kb;
  v16bf a;
#pragma unroll
  for (int i = 0; i < 8; ++i) a[i] = p[i];
#pragma unroll
  for (int i = 0; i < 8; ++i) a[8 + i] = p[8 + i];
  return a;
}

// B-matrix fragment (32x16 bf16) from a K-contiguous ("transposed") LDS tile
// stored as [N][ldk]. lane l: N = l&15 ; K = k0 + (l&16 ? 16 : 0) + i.
__device__ __forceinline__ v16bf load_bT_frag(const bf16_t* s, int ldk,
                                              int col0, int k0, int lane) {
  int n  = lane & 15;
  int kb = k0 + ((lane & 16) ? 16 : 0);
  const bf16_t* p = s + (col0 + n) * ldk + kb;
  v16bf b;
#pragma unroll
  for (int i = 0; i < 16; ++i) b[i] = p[i];
  return b;
}

__device__ __forceinline__ unsigned int pack_bf16(float lo, float hi) {
  union { bf16_t h[2]; unsigned int u; } p;
  p.h[0] = (bf16_t)lo;
  p.h[1] = (bf16_t)hi;
  return p.u;
}

#if HAVE_TDM
// Issue one TDM DMA of `elems` contiguous bf16 (2B) elements from global to
// LDS. 1-D descriptor per CDNA5 ISA (D# groups 0/1).
// Wave-level op (EXEC ignored); caller gates to a single wave and later does
// s_wait_tensorcnt(0) before the workgroup barrier.
__device__ __forceinline__ void tdm_load_1d(const bf16_t* lds_dst,
                                            const bf16_t* gsrc,
                                            unsigned int elems) {
  unsigned long long ga = (unsigned long long)(uintptr_t)gsrc;
  unsigned int lds_off = (unsigned int)(uintptr_t)lds_dst;  // low 32b = LDS byte addr
  u32x4 g0;
  g0[0] = 1u;                                   // count=1, user descriptor
  g0[1] = lds_off;                              // lds_addr
  g0[2] = (unsigned int)ga;                     // global_addr[31:0]
  g0[3] = (unsigned int)(ga >> 32) | (2u << 30);// global_addr[56:32] | type=2
  i32x8 g1;
  g1[0] = 0x10000;                              // data_size=1 (2 bytes)
  g1[1] = (int)((elems & 0xFFFFu) << 16);       // tensor_dim0[15:0]
  g1[2] = (int)(((elems >> 16) & 0xFFFFu) | (1u << 16)); // dim0 hi | tensor_dim1=1
  g1[3] = (int)((elems & 0xFFFFu) << 16);       // tile_dim0 = elems (<=65535)
  g1[4] = 1;                                    // tile_dim1 = 1
  g1[5] = (int)elems;                           // tensor_dim0_stride lo32
  g1[6] = 0;
  g1[7] = 0;
  i32x4 z4 = {0, 0, 0, 0};
  i32x8 z8 = {0, 0, 0, 0, 0, 0, 0, 0};
  // 6-arg form (clang-23 / therock-10.0 headers): groups 0..3 + extra + cpol
  __builtin_amdgcn_tensor_load_to_lds(g0, g1, z4, z4, z8, 0);
}
#endif

// ---------------------------------------------------------------------------
// Kernel 1: QKV = X @ Wqkv with RoPE on Q,K in the epilogue.
// 256 thr (8 waves as 2x4). Block tile 128x128, K-chunks of 32.
// Wave tile 64x32 = 4 A-frags x 2 B-frags = 8 WMMA / K-step.
// ---------------------------------------------------------------------------
__global__ void __launch_bounds__(256)
qkv_gemm_rope(const float* __restrict__ X, const float* __restrict__ W,
              bf16_t* __restrict__ Q, bf16_t* __restrict__ K,
              bf16_t* __restrict__ V) {
  __shared__ bf16_t sA[128 * 32];    // [M][K]
  __shared__ bf16_t sBt[128 * 32];   // [N][K]  (transposed for frag loads)
  const int tid  = threadIdx.x;
  const int lane = tid & 31;
  const int w    = tid >> 5;
  const int wm   = w >> 2;
  const int wn   = w & 3;
  const int mt   = blockIdx.y * 128;
  const int nt   = blockIdx.x * 128;

  v8f acc[4][2];
#pragma unroll
  for (int mi = 0; mi < 4; ++mi)
#pragma unroll
    for (int j = 0; j < 2; ++j) acc[mi][j] = (v8f){};

  for (int k0 = 0; k0 < DMODEL; k0 += 32) {
    // stage A 128x32: 16 f32/thread -> two b128 LDS stores
    {
      int r = tid >> 1, c = (tid & 1) * 16;
      const float* gp = X + (size_t)(mt + r) * DMODEL + k0 + c;
      __builtin_prefetch(gp + 32, 0, 1);
      union { uint4 u[2]; bf16_t h[16]; } st;
#pragma unroll
      for (int i = 0; i < 16; ++i) st.h[i] = (bf16_t)gp[i];
      uint4* sp = (uint4*)&sA[r * 32 + c];
      sp[0] = st.u[0];
      sp[1] = st.u[1];
    }
    // stage B 32x128 transposed: 2 rows x 8 cols per thread -> 8 b32 stores
    {
      int rp = tid >> 4;            // row pair 0..15 -> rows 2rp,2rp+1
      int c  = (tid & 15) * 8;
      const float* g0 = W + (size_t)(k0 + 2 * rp) * QKVDIM + nt + c;
      const float* g1 = g0 + QKVDIM;
      __builtin_prefetch(g0 + (size_t)32 * QKVDIM, 0, 1);
#pragma unroll
      for (int i = 0; i < 8; ++i)
        *(unsigned int*)&sBt[(c + i) * 32 + 2 * rp] = pack_bf16(g0[i], g1[i]);
    }
    __syncthreads();
    v16bf bfrag[2];
#pragma unroll
    for (int j = 0; j < 2; ++j)
      bfrag[j] = load_bT_frag(sBt, 32, wn * 32 + j * 16, 0, lane);
#pragma unroll
    for (int mi = 0; mi < 4; ++mi) {
      v16bf a = load_a_frag(sA, 32, wm * 64 + mi * 16, 0, lane);
#pragma unroll
      for (int j = 0; j < 2; ++j) acc[mi][j] = wmma_bf16(a, bfrag[j], acc[mi][j]);
    }
    __syncthreads();
  }

  // epilogue: RoPE (q,k) + scatter to [B,H,S,hd] bf16
  const int n  = lane & 15;
  const int hi = (lane & 16) ? 8 : 0;
#pragma unroll
  for (int j = 0; j < 2; ++j) {
    const int E    = nt + wn * 32 + j * 16 + n;  // uniform part/head per wave
    const int part = E >> 11;                    // 0=q,1=k,2=v
    const int e    = E & 2047;
    const int h    = e >> 7;
    const int d    = e & 127;
    bf16_t* dst = (part == 0) ? Q : ((part == 1) ? K : V);
    const float inv_freq =
        __expf(-(float)(d & ~1) * 0.07195578492879319f);  // ln(1e4)/128
#pragma unroll
    for (int mi = 0; mi < 4; ++mi) {
#pragma unroll
      for (int r = 0; r < 8; ++r) {
        const int R  = mt + wm * 64 + mi * 16 + r + hi;
        const int bb = R >> 11;
        const int s  = R & 2047;
        float v = acc[mi][j][r];
        float o = v;
        if (part < 2) {
          float pv  = __shfl_xor(v, 1);  // rope partner (even<->odd col)
          float ang = (float)s * inv_freq;
          float sn, cs;
          __sincosf(ang, &sn, &cs);
          o = (d & 1) ? (v * cs + pv * sn) : (v * cs - pv * sn);
        }
        dst[((size_t)(bb * NHEADS + h) * SEQ + s) * HDIM + d] = (bf16_t)o;
      }
    }
  }
}

// ---------------------------------------------------------------------------
// Kernel 2: causal flash attention. 256 thr = 8 waves; 128-query tile.
// Q and K tiles staged by the Tensor Data Mover (async DMA, TENSORcnt);
// V transposed manually into LDS for K-contiguous PV fragments.
// ---------------------------------------------------------------------------
__global__ void __launch_bounds__(256)
flash_attn(const bf16_t* __restrict__ Q, const bf16_t* __restrict__ K,
           const bf16_t* __restrict__ V, bf16_t* __restrict__ Ao) {
  __shared__ bf16_t sQ[128 * HDIM];   // [q][hd]
  __shared__ bf16_t sK[64 * HDIM];    // [key][hd]
  __shared__ bf16_t sVt[HDIM * 64];   // [hd][key]
  __shared__ bf16_t sP[8][16 * 64];   // per-wave P relayout buffer

  const int tid  = threadIdx.x;
  const int lane = tid & 31;
  const int w    = tid >> 5;
  const int qb   = blockIdx.x;
  const int bh   = blockIdx.y;
  const size_t head_off = (size_t)bh * SEQ * HDIM;

  const bf16_t* gQ = Q + head_off + (size_t)qb * 128 * HDIM;
#if HAVE_TDM
  if (w == 0) tdm_load_1d(sQ, gQ, 128 * HDIM);
#else
  {
    const uint4* gq = (const uint4*)gQ;
    uint4* sq = (uint4*)sQ;
#pragma unroll
    for (int i = 0; i < 8; ++i) sq[tid + i * 256] = gq[tid + i * 256];
  }
#endif

  float mrow[8], lrow[8];
  v8f o[8];
#pragma unroll
  for (int r = 0; r < 8; ++r) { mrow[r] = -INFINITY; lrow[r] = 0.f; }
#pragma unroll
  for (int nn = 0; nn < 8; ++nn) o[nn] = (v8f){};

  const float scale = 0.08838834764831845f; // 1/sqrt(128)
  const int n  = lane & 15;
  const int hi = (lane & 16) ? 8 : 0;
  const int kbmax = 2 * qb + 1;

  for (int kb = 0; kb <= kbmax; ++kb) {
    const bf16_t* gK = K + head_off + (size_t)kb * 64 * HDIM;
#if HAVE_TDM
    if (w == 0) tdm_load_1d(sK, gK, 64 * HDIM);
#else
    {
      const uint4* gk = (const uint4*)gK;
      uint4* sk = (uint4*)sK;
#pragma unroll
      for (int i = 0; i < 4; ++i) sk[tid + i * 256] = gk[tid + i * 256];
    }
#endif
    // stage V tile transposed -> sVt[hd][key]; 2 keys x 16 hd per thread,
    // packed pair stores (b32).
    {
      int kp = tid >> 3;            // key pair 0..31 -> keys 2kp,2kp+1
      int hc = (tid & 7) * 16;
      const uint4* gv0 = (const uint4*)(V + head_off + (size_t)kb * 64 * HDIM +
                                        (size_t)(2 * kp) * HDIM + hc);
      const uint4* gv1 = (const uint4*)((const bf16_t*)gv0 + HDIM);
      union { uint4 u[2]; bf16_t h[16]; } r0, r1;
      r0.u[0] = gv0[0]; r0.u[1] = gv0[1];
      r1.u[0] = gv1[0]; r1.u[1] = gv1[1];
#pragma unroll
      for (int i = 0; i < 16; ++i) {
        union { bf16_t h[2]; unsigned int u; } p;
        p.h[0] = r0.h[i];
        p.h[1] = r1.h[i];
        *(unsigned int*)&sVt[(hc + i) * 64 + 2 * kp] = p.u;
      }
    }
#if HAVE_TDM
    if (w == 0) __builtin_amdgcn_s_wait_tensorcnt(0);
#endif
    __syncthreads();

    // scores = Q @ K^T  (16x64 per wave)
    v8f sc[4];
#pragma unroll
    for (int t = 0; t < 4; ++t) sc[t] = (v8f){};
#pragma unroll
    for (int kc = 0; kc < 4; ++kc) {
      v16bf aq = load_a_frag(sQ, HDIM, w * 16, kc * 32, lane);
#pragma unroll
      for (int t = 0; t < 4; ++t) {
        v16bf bk = load_bT_frag(sK, HDIM, t * 16, kc * 32, lane);
        sc[t] = wmma_bf16(aq, bk, sc[t]);
      }
    }

    // scale, causal mask (only last two key blocks partial), row-max
    const bool maskblk = (kb >= 2 * qb);
    float bmax[8];
#pragma unroll
    for (int r = 0; r < 8; ++r) bmax[r] = -INFINITY;
#pragma unroll
    for (int t = 0; t < 4; ++t)
#pragma unroll
      for (int r = 0; r < 8; ++r) {
        float v = sc[t][r] * scale;
        if (maskblk) {
          int gq_ = qb * 128 + w * 16 + r + hi;
          int gk_ = kb * 64 + t * 16 + n;
          if (gk_ > gq_) v = -INFINITY;
        }
        sc[t][r] = v;
        bmax[r] = fmaxf(bmax[r], v);
      }
#pragma unroll
    for (int r = 0; r < 8; ++r) {
      bmax[r] = fmaxf(bmax[r], __shfl_xor(bmax[r], 1));
      bmax[r] = fmaxf(bmax[r], __shfl_xor(bmax[r], 2));
      bmax[r] = fmaxf(bmax[r], __shfl_xor(bmax[r], 4));
      bmax[r] = fmaxf(bmax[r], __shfl_xor(bmax[r], 8));
    }

    // online softmax state update
    float sc_old[8], bsum[8];
#pragma unroll
    for (int r = 0; r < 8; ++r) {
      float mn = fmaxf(mrow[r], bmax[r]);
      sc_old[r] = __expf(mrow[r] - mn);
      mrow[r] = mn;
      lrow[r] *= sc_old[r];
      bsum[r] = 0.f;
    }
#pragma unroll
    for (int t = 0; t < 4; ++t)
#pragma unroll
      for (int r = 0; r < 8; ++r) {
        float p = __expf(sc[t][r] - mrow[r]);
        sc[t][r] = p;
        bsum[r] += p;
      }
#pragma unroll
    for (int r = 0; r < 8; ++r) {
      bsum[r] += __shfl_xor(bsum[r], 1);
      bsum[r] += __shfl_xor(bsum[r], 2);
      bsum[r] += __shfl_xor(bsum[r], 4);
      bsum[r] += __shfl_xor(bsum[r], 8);
      lrow[r] += bsum[r];
    }
#pragma unroll
    for (int nn = 0; nn < 8; ++nn)
#pragma unroll
      for (int r = 0; r < 8; ++r) o[nn][r] *= sc_old[r];

    // P (C-layout) -> per-wave LDS -> A-layout fragments
    bf16_t* pw = sP[w];
#pragma unroll
    for (int t = 0; t < 4; ++t)
#pragma unroll
      for (int r = 0; r < 8; ++r)
        pw[(r + hi) * 64 + t * 16 + n] = (bf16_t)sc[t][r];

    // O += P @ V  (16x128 per wave)
#pragma unroll
    for (int kc2 = 0; kc2 < 2; ++kc2) {
      v16bf ap = load_a_frag(pw, 64, 0, kc2 * 32, lane);
#pragma unroll
      for (int nn = 0; nn < 8; ++nn) {
        v16bf bv = load_bT_frag(sVt, 64, nn * 16, kc2 * 32, lane);
        o[nn] = wmma_bf16(ap, bv, o[nn]);
      }
    }
    __syncthreads();
  }

  // normalize + store bf16 attn output as [B,S,D] (D = h*128 + col)
  const int b = bh >> 4, h = bh & 15;
#pragma unroll
  for (int r = 0; r < 8; ++r) {
    const float il = 1.0f / lrow[r];
    const int srow = qb * 128 + w * 16 + r + hi;
#pragma unroll
    for (int nn = 0; nn < 8; ++nn) {
      Ao[((size_t)b * SEQ + srow) * DMODEL + h * HDIM + nn * 16 + n] =
          (bf16_t)(o[nn][r] * il);
    }
  }
}

// ---------------------------------------------------------------------------
// Kernel 3: out = attn @ Wo. attn:[4096,2048] bf16, Wo:[2048,2048] f32.
// ---------------------------------------------------------------------------
__global__ void __launch_bounds__(256)
out_gemm(const bf16_t* __restrict__ A, const float* __restrict__ W,
         float* __restrict__ Out) {
  __shared__ bf16_t sA[128 * 32];
  __shared__ bf16_t sBt[128 * 32];
  const int tid  = threadIdx.x;
  const int lane = tid & 31;
  const int w    = tid >> 5;
  const int wm   = w >> 2;
  const int wn   = w & 3;
  const int mt   = blockIdx.y * 128;
  const int nt   = blockIdx.x * 128;

  v8f acc[4][2];
#pragma unroll
  for (int mi = 0; mi < 4; ++mi)
#pragma unroll
    for (int j = 0; j < 2; ++j) acc[mi][j] = (v8f){};

  for (int k0 = 0; k0 < DMODEL; k0 += 32) {
    // stage A 128x32 bf16 (2 uint4 copies per thread)
    {
      int r = tid >> 1, c = (tid & 1) * 16;
      const uint4* gp = (const uint4*)(A + (size_t)(mt + r) * DMODEL + k0 + c);
      __builtin_prefetch(gp + 4, 0, 1);
      uint4* sp = (uint4*)&sA[r * 32 + c];
      sp[0] = gp[0];
      sp[1] = gp[1];
    }
    // stage B 32x128 f32 -> transposed bf16, paired rows -> b32 stores
    {
      int rp = tid >> 4;
      int c  = (tid & 15) * 8;
      const float* g0 = W + (size_t)(k0 + 2 * rp) * DMODEL + nt + c;
      const float* g1 = g0 + DMODEL;
      __builtin_prefetch(g0 + (size_t)32 * DMODEL, 0, 1);
#pragma unroll
      for (int i = 0; i < 8; ++i)
        *(unsigned int*)&sBt[(c + i) * 32 + 2 * rp] = pack_bf16(g0[i], g1[i]);
    }
    __syncthreads();
    v16bf bfrag[2];
#pragma unroll
    for (int j = 0; j < 2; ++j)
      bfrag[j] = load_bT_frag(sBt, 32, wn * 32 + j * 16, 0, lane);
#pragma unroll
    for (int mi = 0; mi < 4; ++mi) {
      v16bf a = load_a_frag(sA, 32, wm * 64 + mi * 16, 0, lane);
#pragma unroll
      for (int j = 0; j < 2; ++j) acc[mi][j] = wmma_bf16(a, bfrag[j], acc[mi][j]);
    }
    __syncthreads();
  }

  const int n  = lane & 15;
  const int hi = (lane & 16) ? 8 : 0;
#pragma unroll
  for (int j = 0; j < 2; ++j) {
    const int ct = nt + wn * 32 + j * 16 + n;
#pragma unroll
    for (int mi = 0; mi < 4; ++mi) {
#pragma unroll
      for (int r = 0; r < 8; ++r) {
        const int R = mt + wm * 64 + mi * 16 + r + hi;
        Out[(size_t)R * DMODEL + ct] = acc[mi][j][r];
      }
    }
  }
}

// ---------------------------------------------------------------------------
extern "C" void kernel_launch(void* const* d_in, const int* in_sizes, int n_in,
                              void* d_out, int out_size, void* d_ws,
                              size_t ws_size, hipStream_t stream) {
  (void)in_sizes; (void)n_in; (void)out_size; (void)ws_size;
  const float* x    = (const float*)d_in[0];
  const float* wqkv = (const float*)d_in[1];
  const float* wo   = (const float*)d_in[2];
  float* out = (float*)d_out;

  const size_t elems = (size_t)BATCH * SEQ * DMODEL;  // 8,388,608 per tensor
  bf16_t* q  = (bf16_t*)d_ws;
  bf16_t* k  = q + elems;
  bf16_t* v  = k + elems;
  bf16_t* ao = v + elems;

  qkv_gemm_rope<<<dim3(QKVDIM / 128, MROWS / 128), 256, 0, stream>>>(
      x, wqkv, q, k, v);
  flash_attn<<<dim3(SEQ / 128, BATCH * NHEADS), 256, 0, stream>>>(q, k, v, ao);
  out_gemm<<<dim3(DMODEL / 128, MROWS / 128), 256, 0, stream>>>(ao, wo, out);
}
